// H2GCN_90675349553252
// MI455X (gfx1250) — matrix operand order repaired and checked by reading, not compiled, not run on hardware
//
#include <hip/hip_runtime.h>
#include <hip/hip_bf16.h>

// ---------------------------------------------------------------------------
// H2GCN forward for MI455X (gfx1250, wave32, WMMA f32_16x16x32_f16).
//
// r0 = relu(x @ w_embed)                       [N,64]
// r1 = [spmm(a1,r0), spmm(a2,r0)]              [N,128]
// r2 = [spmm(a1,r1), spmm(a2,r1)]              [N,256]
// out = log_softmax([r2,r0,r1] @ w_classify)   [N,64]
//
// Workspace: rf[N][448] f32 (cols [0:256)=r2, [256:320)=r0, [320:448)=r1,
// i.e. the reference concat order), plus f16 A/B operand buffers pre-packed
// into the exact WMMA lane layout so the GEMM hot loops are pure
// b128-load + v_wmma.
// ---------------------------------------------------------------------------

typedef __attribute__((ext_vector_type(16))) _Float16 v16h;
typedef __attribute__((ext_vector_type(8)))  float    v8f;

#define STRIDE 448   // r_final width
#define CT     4     // col tiles (64 cols / 16)

// ---------------------------------------------------------------------------
__global__ void h2gcn_zero_ws(float4* __restrict__ p, size_t n4) {
  size_t i = (size_t)blockIdx.x * blockDim.x + threadIdx.x;
  size_t stride = (size_t)gridDim.x * blockDim.x;
  float4 z = {0.f, 0.f, 0.f, 0.f};
  for (; i < n4; i += stride) p[i] = z;
}

// ---------------------------------------------------------------------------
// Pack a row-major f32 matrix [rows, K] into WMMA-A f16 layout:
// dst[((rowTile*steps + step)*32 + lane)*16 + e], where for lane = half*16+m:
//   e in [0,8):  A[rowTile*16+m][step*32 + half*8 + e]
//   e in [8,16): A[rowTile*16+m][step*32 + half*8 + 16 + (e-8)]
// One wave per (rowTile, step) unit.
__global__ void h2gcn_pack_a(const float* __restrict__ src, _Float16* __restrict__ dst,
                             int K, int rowTiles, int steps) {
  const int lane   = threadIdx.x & 31;
  const int nUnits = rowTiles * steps;
  const int uStep  = (int)(((size_t)gridDim.x * blockDim.x) >> 5);
  int unit = (int)(((size_t)blockIdx.x * blockDim.x + threadIdx.x) >> 5);
  const int half = lane >> 4, m = lane & 15;
  for (; unit < nUnits; unit += uStep) {
    const int rowTile = unit / steps;
    const int step    = unit % steps;
    const float* ar = src + (size_t)(rowTile * 16 + m) * K + step * 32 + half * 8;
    v16h a;
#pragma unroll
    for (int e = 0; e < 8; ++e) {
      a[e]     = (_Float16)ar[e];
      a[e + 8] = (_Float16)ar[16 + e];
    }
    *(v16h*)(dst + ((size_t)unit * 32 + lane) * 16) = a;
  }
}

// ---------------------------------------------------------------------------
// Pack w [K,64] f32 into WMMA-B f16 layout:
// dst[((step*CT + ct)*32 + lane)*16 + e] = w[step*32 + half*8 + (e<8 ? e : e+8)][ct*16 + m]
__global__ void h2gcn_pack_b(const float* __restrict__ w, _Float16* __restrict__ dst,
                             int steps) {
  const int lane   = threadIdx.x & 31;
  const int nUnits = steps * CT;
  const int uStep  = (int)(((size_t)gridDim.x * blockDim.x) >> 5);
  int unit = (int)(((size_t)blockIdx.x * blockDim.x + threadIdx.x) >> 5);
  const int half = lane >> 4, m = lane & 15;
  for (; unit < nUnits; unit += uStep) {
    const int step = unit >> 2;
    const int ct   = unit & 3;
    const float* wc = w + (size_t)(step * 32 + half * 8) * 64 + ct * 16 + m;
    v16h b;
#pragma unroll
    for (int e = 0; e < 16; ++e) b[e] = (_Float16)wc[(size_t)(e < 8 ? e : e + 8) * 64];
    *(v16h*)(dst + ((size_t)unit * 32 + lane) * 16) = b;
  }
}

// ---------------------------------------------------------------------------
// r0 = relu(xA @ wB) -> rf[:, 256:320].  One wave per 16-row tile, all 4 col
// tiles (4 accumulators).  Hot loop: 1 A load + 4 B loads + 4 v_wmma.
__global__ void __launch_bounds__(128)
h2gcn_embed_gemm(const _Float16* __restrict__ xA, const _Float16* __restrict__ wB,
                 float* __restrict__ rf, int rowTiles) {
  const int lane    = threadIdx.x & 31;
  const int rowTile = blockIdx.x * 4 + (threadIdx.x >> 5);
  if (rowTile >= rowTiles) return;
  const int half = lane >> 4, m = lane & 15;
  const int steps = 512 / 32;

  v8f acc0 = {}, acc1 = {}, acc2 = {}, acc3 = {};
  const _Float16* aP = xA + (size_t)rowTile * steps * 512 + (size_t)lane * 16;
  const _Float16* bP = wB + (size_t)lane * 16;
  for (int s = 0; s < steps; ++s) {
    v16h a  = *(const v16h*)aP;
    v16h b0 = *(const v16h*)(bP);
    v16h b1 = *(const v16h*)(bP + 512);
    v16h b2 = *(const v16h*)(bP + 1024);
    v16h b3 = *(const v16h*)(bP + 1536);
    acc0 = __builtin_amdgcn_wmma_f32_16x16x32_f16(false, a, false, b0, (short)0, acc0, false, false);
    acc1 = __builtin_amdgcn_wmma_f32_16x16x32_f16(false, a, false, b1, (short)0, acc1, false, false);
    acc2 = __builtin_amdgcn_wmma_f32_16x16x32_f16(false, a, false, b2, (short)0, acc2, false, false);
    acc3 = __builtin_amdgcn_wmma_f32_16x16x32_f16(false, a, false, b3, (short)0, acc3, false, false);
    aP += 512;
    bP += 2048;
  }
  // D layout: elem v -> row = rowTile*16 + half*8 + v, col = ct*16 + m.
#pragma unroll
  for (int v = 0; v < 8; ++v) {
    float* orow = rf + (size_t)(rowTile * 16 + half * 8 + v) * STRIDE + 256 + m;
    float r0 = acc0[v], r1 = acc1[v], r2 = acc2[v], r3 = acc3[v];
    orow[0]  = r0 > 0.f ? r0 : 0.f;
    orow[16] = r1 > 0.f ? r1 : 0.f;
    orow[32] = r2 > 0.f ? r2 : 0.f;
    orow[48] = r3 > 0.f ? r3 : 0.f;
  }
}

// ---------------------------------------------------------------------------
// COO SpMM scatter: hout[dst, c] += val * hin[src, c], c in [0, F).
// One wave per edge; prefetch the next edge's source row (global_prefetch_b8)
// to hide the random-gather latency.
__global__ void h2gcn_spmm_atomic(const int* __restrict__ dstIdx,
                                  const int* __restrict__ srcIdx,
                                  const float* __restrict__ val,
                                  int nEdges,
                                  const float* __restrict__ hin,
                                  float* __restrict__ hout,
                                  int F) {
  const int lane   = threadIdx.x & 31;
  const int gwave  = (int)(((size_t)blockIdx.x * blockDim.x + threadIdx.x) >> 5);
  const int nwaves = (int)(((size_t)gridDim.x * blockDim.x) >> 5);
  const int fpl    = F >> 5;  // floats per lane
  for (int e = gwave; e < nEdges; e += nwaves) {
    const int eN = e + nwaves;
    if (eN < nEdges) {
      const float* pf = hin + (size_t)srcIdx[eN] * STRIDE + lane * fpl;
      __builtin_prefetch(pf, 0, 1);
    }
    const int   d  = dstIdx[e];
    const int   s  = srcIdx[e];
    const float wv = val[e];
    const float* hs = hin + (size_t)s * STRIDE;
    float*       hd = hout + (size_t)d * STRIDE;
    for (int c = lane; c < F; c += 32) {
      atomicAdd(&hd[c], wv * hs[c]);
    }
  }
}

// ---------------------------------------------------------------------------
// out = log_softmax(rfA @ wB).  One wave per row tile x 4 col tiles into a
// per-wave 16x64 LDS slab, then one thread per row does the 64-wide
// log-softmax and writes the row.
__global__ void __launch_bounds__(128)
h2gcn_classify_lsm(const _Float16* __restrict__ rfA, const _Float16* __restrict__ wB,
                   float* __restrict__ out, int rowTiles) {
  __shared__ float lds[4][16][64];
  const int lane = threadIdx.x & 31;
  const int wv   = threadIdx.x >> 5;
  const int rowTile = blockIdx.x * 4 + wv;
  const int half = lane >> 4, m = lane & 15;
  const int steps = STRIDE / 32;  // 14

  if (rowTile < rowTiles) {
    v8f acc0 = {}, acc1 = {}, acc2 = {}, acc3 = {};
    const _Float16* aP = rfA + (size_t)rowTile * steps * 512 + (size_t)lane * 16;
    const _Float16* bP = wB + (size_t)lane * 16;
    for (int s = 0; s < steps; ++s) {
      v16h a  = *(const v16h*)aP;
      v16h b0 = *(const v16h*)(bP);
      v16h b1 = *(const v16h*)(bP + 512);
      v16h b2 = *(const v16h*)(bP + 1024);
      v16h b3 = *(const v16h*)(bP + 1536);
      acc0 = __builtin_amdgcn_wmma_f32_16x16x32_f16(false, a, false, b0, (short)0, acc0, false, false);
      acc1 = __builtin_amdgcn_wmma_f32_16x16x32_f16(false, a, false, b1, (short)0, acc1, false, false);
      acc2 = __builtin_amdgcn_wmma_f32_16x16x32_f16(false, a, false, b2, (short)0, acc2, false, false);
      acc3 = __builtin_amdgcn_wmma_f32_16x16x32_f16(false, a, false, b3, (short)0, acc3, false, false);
      aP += 512;
      bP += 2048;
    }
#pragma unroll
    for (int v = 0; v < 8; ++v) {
      const int r = half * 8 + v;
      lds[wv][r][m]      = acc0[v];
      lds[wv][r][16 + m] = acc1[v];
      lds[wv][r][32 + m] = acc2[v];
      lds[wv][r][48 + m] = acc3[v];
    }
  }
  __syncthreads();

  const int r = threadIdx.x;           // 0..127; rows 0..63 valid
  const int grow = blockIdx.x * 64 + r;
  if (r < 64 && grow < rowTiles * 16) {
    const float* row = lds[r >> 4][r & 15];
    float mx = -3.402823466e38f;
    for (int c = 0; c < 64; ++c) mx = fmaxf(mx, row[c]);
    float sum = 0.f;
    for (int c = 0; c < 64; ++c) sum += expf(row[c] - mx);
    const float lse = mx + logf(sum);
    float* o = out + (size_t)grow * 64;
    for (int c = 0; c < 64; ++c) o[c] = row[c] - lse;
  }
}

// ---------------------------------------------------------------------------
static inline size_t align256(size_t x) { return (x + 255) & ~(size_t)255; }

extern "C" void kernel_launch(void* const* d_in, const int* in_sizes, int n_in,
                              void* d_out, int out_size, void* d_ws, size_t ws_size,
                              hipStream_t stream) {
  const float* x      = (const float*)d_in[0];  // [N,512]
  const int*   a1_idx = (const int*)  d_in[1];  // [2,E1]
  const float* a1_val = (const float*)d_in[2];  // [E1]
  const int*   a2_idx = (const int*)  d_in[3];  // [2,E2]
  const float* a2_val = (const float*)d_in[4];  // [E2]
  const float* w_emb  = (const float*)d_in[5];  // [512,64]
  const float* w_cls  = (const float*)d_in[6];  // [448,64]
  float* out = (float*)d_out;

  const int N  = in_sizes[0] / 512;
  const int E1 = in_sizes[1] / 2;
  const int E2 = in_sizes[3] / 2;
  const int rowTiles = N / 16;        // N = 100000 -> 6250
  const int gemmGrid = (rowTiles + 3) / 4;

  // Workspace carve-up (256B-aligned sub-buffers).
  char* ws = (char*)d_ws;
  size_t off = 0;
  float* rf = (float*)(ws + off);               off += align256((size_t)N * STRIDE * 4);
  _Float16* xA  = (_Float16*)(ws + off);        off += align256((size_t)N * 512 * 2);
  _Float16* rfA = (_Float16*)(ws + off);        off += align256((size_t)N * STRIDE * 2);
  _Float16* wB0 = (_Float16*)(ws + off);        off += align256((size_t)512 * 64 * 2);
  _Float16* wB1 = (_Float16*)(ws + off);        off += align256((size_t)448 * 64 * 2);
  (void)ws_size;

  // 0) zero the atomic-accumulation buffer (harness poisons ws).
  h2gcn_zero_ws<<<2048, 256, 0, stream>>>((float4*)rf, (size_t)N * STRIDE / 4);

  // 1) pack operands into WMMA-native f16 layouts.
  h2gcn_pack_b<<<8, 256, 0, stream>>>(w_emb, wB0, 512 / 32);
  h2gcn_pack_b<<<8, 256, 0, stream>>>(w_cls, wB1, STRIDE / 32);
  h2gcn_pack_a<<<2048, 256, 0, stream>>>(x, xA, 512, rowTiles, 512 / 32);

  // 2) r0 = relu(x @ w_embed) -> rf[:,256:320]
  h2gcn_embed_gemm<<<gemmGrid, 128, 0, stream>>>(xA, wB0, rf, rowTiles);

  // 3) r1 = [spmm(a1,r0), spmm(a2,r0)] -> rf[:,320:384], rf[:,384:448]
  h2gcn_spmm_atomic<<<4096, 256, 0, stream>>>(a1_idx, a1_idx + E1, a1_val, E1,
                                              rf + 256, rf + 320, 64);
  h2gcn_spmm_atomic<<<8192, 256, 0, stream>>>(a2_idx, a2_idx + E2, a2_val, E2,
                                              rf + 256, rf + 384, 64);

  // 4) r2 = [spmm(a1,r1), spmm(a2,r1)] -> rf[:,0:128], rf[:,128:256]
  h2gcn_spmm_atomic<<<4096, 256, 0, stream>>>(a1_idx, a1_idx + E1, a1_val, E1,
                                              rf + 320, rf + 0, 128);
  h2gcn_spmm_atomic<<<8192, 256, 0, stream>>>(a2_idx, a2_idx + E2, a2_val, E2,
                                              rf + 320, rf + 128, 128);

  // 5) pack r_final -> f16 WMMA-A layout, then classify + log_softmax.
  h2gcn_pack_a<<<2048, 256, 0, stream>>>(rf, rfA, STRIDE, rowTiles, STRIDE / 32);
  h2gcn_classify_lsm<<<gemmGrid, 128, 0, stream>>>(rfA, wB1, out, rowTiles);
}
